// GNN_53249004536466
// MI455X (gfx1250) — matrix operand order repaired and checked by reading, not compiled, not run on hardware
//
#include <hip/hip_runtime.h>
#include <hip/hip_bf16.h>
#include <stdint.h>

typedef __attribute__((ext_vector_type(16))) _Float16 v16h;
typedef __attribute__((ext_vector_type(8)))  float    v8f;

#define GCN_DIN 128
#define GCN_DH  32
#define GCN_DOUT 2

// ---------------- degree / normalization ----------------
__global__ void deg_init_kernel(float* __restrict__ deg, int n) {
    int i = blockIdx.x * 256 + threadIdx.x;
    if (i < n) deg[i] = 1.0f;   // self-loop contributes 1
}

__global__ void deg_count_kernel(const long long* __restrict__ dst,
                                 float* __restrict__ deg, int e) {
    int i = blockIdx.x * 256 + threadIdx.x;
    if (i < e) atomicAdd(&deg[(int)dst[i]], 1.0f);
}

__global__ void deg_rsqrt_kernel(float* __restrict__ deg, int n) {
    int i = blockIdx.x * 256 + threadIdx.x;
    if (i < n) deg[i] = rsqrtf(deg[i]);   // deg >= 1 always
}

// ---------------- layer-1 transform: m = x @ W1 via WMMA ----------------
// Block = 256 threads = 8 waves; each wave computes a 16-row x 32-col tile.
// B fragments are pre-packed into LDS in the exact per-lane WMMA register
// layout, so each fragment load is 2x ds_load_b128 (conflict-free).
// Also fuses the self-loop term: agg = c * dinv[row]^2.
__global__ __launch_bounds__(256) void gemm1_wmma_kernel(
        const float* __restrict__ x, const float* __restrict__ W1,
        const float* __restrict__ dinv,
        float* __restrict__ m, float* __restrict__ agg, int n) {
    // bfrag[kstep][ntile][lane] = v16h B fragment (8 KB total)
    __shared__ v16h bfrag[4][2][32];
    {
        // 4*2*32 = 256 fragments -> exactly one per thread
        int t     = threadIdx.x;
        int kstep = t >> 6;          // 0..3
        int ntile = (t >> 5) & 1;    // 0..1
        int lane  = t & 31;
        // B layout used here (contiguous k, same as A): col = lane&15,
        //  lanes 0-15 hold K = k0+0..15, lanes 16-31 hold K = k0+16..31
        int kB  = kstep * 32 + (lane >> 4) * 16;
        int col = (lane & 15) + ntile * 16;
        v16h tmp;
        #pragma unroll
        for (int j = 0; j < 16; ++j)
            tmp[j] = (_Float16)W1[(kB + j) * GCN_DH + col];
        bfrag[kstep][ntile][lane] = tmp;
    }
    __syncthreads();

    int wave = threadIdx.x >> 5;
    int lane = threadIdx.x & 31;
    int rowBase = blockIdx.x * 128 + wave * 16;
    if (rowBase + 16 > n) return;          // wave-uniform; n % 16 == 0

    int mrow   = lane & 15;                // A: row M; B/C: column N
    int hiHalf = lane >> 4;                // 0 or 1
    const float* xrow = x + (size_t)(rowBase + mrow) * GCN_DIN;

    v8f c0 = {}; v8f c1 = {};
    #pragma unroll
    for (int kstep = 0; kstep < 4; ++kstep) {
        int k0 = kstep * 32;
        // A fragment (contiguous k order, same as B):
        //  lanes 0-15 : K = k0+0..15 (elems 0-15)
        //  lanes 16-31: K = k0+16..31
        v16h a;
        int kA = k0 + hiHalf * 16;
        #pragma unroll
        for (int j = 0; j < 8; ++j)
            a[j]     = (_Float16)__builtin_nontemporal_load(xrow + kA + j);
        #pragma unroll
        for (int j = 0; j < 8; ++j)
            a[8 + j] = (_Float16)__builtin_nontemporal_load(xrow + kA + 8 + j);

        v16h b0  = bfrag[kstep][0][lane];
        v16h b1v = bfrag[kstep][1][lane];
        c0 = __builtin_amdgcn_wmma_f32_16x16x32_f16(false, a, false, b0,
                                                    (short)0, c0, false, false);
        c1 = __builtin_amdgcn_wmma_f32_16x16x32_f16(false, a, false, b1v,
                                                    (short)0, c1, false, false);
    }
    // C layout: VGPR j -> row = rowBase + j + 8*hiHalf, col = mrow (+16)
    #pragma unroll
    for (int j = 0; j < 8; ++j) {
        int r = rowBase + j + 8 * hiHalf;
        float w  = dinv[r];
        float w2 = w * w;
        m  [(size_t)r * GCN_DH + mrow]      = c0[j];
        m  [(size_t)r * GCN_DH + mrow + 16] = c1[j];
        agg[(size_t)r * GCN_DH + mrow]      = c0[j] * w2;   // self-loop init
        agg[(size_t)r * GCN_DH + mrow + 16] = c1[j] * w2;
    }
}

// ---------------- layer-1 edge scatter ----------------
// 8 threads per edge, float4 per thread, fp32 atomics (L2-resident target).
__global__ void scatter1_kernel(const long long* __restrict__ src,
                                const long long* __restrict__ dst,
                                const float* __restrict__ dinv,
                                const float* __restrict__ m,
                                float* __restrict__ agg, int e) {
    int tid  = blockIdx.x * 256 + threadIdx.x;
    int edge = tid >> 3;
    int part = tid & 7;
    if (edge >= e) return;
    int s = (int)src[edge];
    int d = (int)dst[edge];
    float w = dinv[s] * dinv[d];
    const float4 v = *(const float4*)(m + (size_t)s * GCN_DH + part * 4);
    float* o = agg + (size_t)d * GCN_DH + part * 4;
    atomicAdd(o + 0, v.x * w);
    atomicAdd(o + 1, v.y * w);
    atomicAdd(o + 2, v.z * w);
    atomicAdd(o + 3, v.w * w);
}

// ---------------- fused bias + ReLU + layer-2 transform + out init ------
__global__ void layer2_kernel(const float* __restrict__ agg,
                              const float* __restrict__ b1,
                              const float* __restrict__ W2,
                              const float* __restrict__ dinv,
                              const float* __restrict__ b2,
                              float* __restrict__ m2,
                              float* __restrict__ out, int n) {
    int i = blockIdx.x * 256 + threadIdx.x;
    if (i >= n) return;
    const float* row = agg + (size_t)i * GCN_DH;
    float a0 = 0.0f, a1 = 0.0f;
    #pragma unroll
    for (int j = 0; j < GCN_DH; ++j) {
        float h = fmaxf(row[j] + b1[j], 0.0f);
        a0 += h * W2[j * GCN_DOUT + 0];
        a1 += h * W2[j * GCN_DOUT + 1];
    }
    m2[(size_t)i * GCN_DOUT + 0] = a0;
    m2[(size_t)i * GCN_DOUT + 1] = a1;
    float w  = dinv[i];
    float w2 = w * w;
    out[(size_t)i * GCN_DOUT + 0] = a0 * w2 + b2[0];   // self-loop + bias
    out[(size_t)i * GCN_DOUT + 1] = a1 * w2 + b2[1];
}

// ---------------- layer-2 edge scatter ----------------
__global__ void scatter2_kernel(const long long* __restrict__ src,
                                const long long* __restrict__ dst,
                                const float* __restrict__ dinv,
                                const float* __restrict__ m2,
                                float* __restrict__ out, int e) {
    int i = blockIdx.x * 256 + threadIdx.x;
    if (i >= e) return;
    int s = (int)src[i];
    int d = (int)dst[i];
    float w = dinv[s] * dinv[d];
    float2 v = *(const float2*)(m2 + (size_t)s * GCN_DOUT);
    atomicAdd(&out[(size_t)d * GCN_DOUT + 0], v.x * w);
    atomicAdd(&out[(size_t)d * GCN_DOUT + 1], v.y * w);
}

extern "C" void kernel_launch(void* const* d_in, const int* in_sizes, int n_in,
                              void* d_out, int out_size, void* d_ws, size_t ws_size,
                              hipStream_t stream) {
    const float*     x   = (const float*)d_in[0];
    const long long* ei  = (const long long*)d_in[1];   // int64 (2, E)
    const float*     W1  = (const float*)d_in[2];
    const float*     b1  = (const float*)d_in[3];
    const float*     W2  = (const float*)d_in[4];
    const float*     b2  = (const float*)d_in[5];

    int n = in_sizes[0] / GCN_DIN;   // 100000
    int e = in_sizes[1] / 2;         // 3200000
    const long long* src = ei;
    const long long* dst = ei + e;

    char* ws = (char*)d_ws;
    float* dinv = (float*)ws;                          // n floats (deg -> dinv)
    float* m    = (float*)(ws + (size_t)(1u  << 20));  // n*32 floats (12.8 MB)
    float* agg  = (float*)(ws + (size_t)(16u << 20));  // n*32 floats (12.8 MB)
    float* m2   = (float*)(ws + (size_t)(32u << 20));  // n*2 floats  (0.8 MB)
    float* out  = (float*)d_out;

    int gN    = (n + 255) / 256;
    int gE    = (e + 255) / 256;
    int gRows = (n + 127) / 128;
    int gScat = (e * 8 + 255) / 256;

    deg_init_kernel <<<gN, 256, 0, stream>>>(dinv, n);
    deg_count_kernel<<<gE, 256, 0, stream>>>(dst, dinv, e);
    deg_rsqrt_kernel<<<gN, 256, 0, stream>>>(dinv, n);

    gemm1_wmma_kernel<<<gRows, 256, 0, stream>>>(x, W1, dinv, m, agg, n);
    scatter1_kernel  <<<gScat, 256, 0, stream>>>(src, dst, dinv, m, agg, e);

    layer2_kernel  <<<gN, 256, 0, stream>>>(agg, b1, W2, dinv, b2, m2, out, n);
    scatter2_kernel<<<gE, 256, 0, stream>>>(src, dst, dinv, m2, out, e);
}
